// UpdateFunctionForEventLink_592705487030
// MI455X (gfx1250) — matrix operand order repaired and checked by reading, not compiled, not run on hardware
//
#include <hip/hip_runtime.h>

typedef _Float16 f16;
typedef __attribute__((ext_vector_type(16))) _Float16 v16h;
typedef __attribute__((ext_vector_type(8)))  _Float16 f16x8;
typedef __attribute__((ext_vector_type(8)))  float    v8f;
typedef __attribute__((ext_vector_type(4)))  float    f32x4;

#define HDIM 512
#define NDIM 8192
#define BATCH 8
#define BM 64
#define BN 64
#define BK 32
#define KSTEPS (HDIM / BK)     // 16
#define PB 40                  // B LDS pitch in f16 (32 data + 8 pad) -> 80B rows, 16B aligned
#define BBUF (2 * BN * PB)     // one B buffer in f16 elements (5120)

union V16U { v16h v; f16x8 h[2]; };

// ---------------------------------------------------------------------------
// Pre-pass: convert W_ih / W_hh fp32 -> f16 into d_ws, in WMMA-A-fragment-ready
// swizzled order: chunk(s, mtile, kstep, lane) -> 32 bytes; out offset = tid*16B.
// Lane layout (contiguous k): m = lane&15, kg = lane>>4; half0: K = 16kg..+7,
// half1: K = 16kg+8..+7 (within the 32-wide k-step).
// ---------------------------------------------------------------------------
__global__ __launch_bounds__(256) void convert_weights_kernel(
    const float* __restrict__ W_ih, const float* __restrict__ W_hh,
    f16* __restrict__ Wc)
{
    int t = blockIdx.x * 256 + threadIdx.x;   // [0, 196608)
    int h  = t & 1;
    int l  = (t >> 1) & 31;
    int ks = (t >> 6) & 15;
    int mt = (t >> 10) & 31;
    int s  = t >> 15;                          // 0..5
    const float* W = (s < 3) ? W_ih : W_hh;
    int gate = (s < 3) ? s : s - 3;
    int m = l & 15, kg = l >> 4;
    int row = gate * HDIM + mt * 16 + m;
    int kb  = ks * BK + kg * 16 + h * 8;
    const float* src = W + (size_t)row * HDIM + kb;
    f16x8 o;
#pragma unroll
    for (int j = 0; j < 8; ++j) o[j] = (f16)src[j];
    *(f16x8*)(Wc + (size_t)t * 8) = o;
}

// ---------------------------------------------------------------------------
// Fused dual-GEMM + GRU gates. A fragments stream from L2-resident f16 weights;
// B tiles (m_wv, e_wv) are double-buffered through LDS with an f16 transpose.
// ---------------------------------------------------------------------------
__global__ __launch_bounds__(256) void gru_fused_kernel(
    const float* __restrict__ e_wv, const float* __restrict__ m_wv,
    const f16*  __restrict__ Wc,
    float* __restrict__ out)
{
    __shared__ __align__(16) f16 Bs[2 * BBUF];   // 20 KB, double buffered

    const int tid    = threadIdx.x;
    const int lane   = tid & 31;
    const int wid    = tid >> 5;
    const int wave_m = wid >> 1;   // 0..3
    const int wave_n = wid & 1;    // 0..1
    const int lhalf  = lane >> 4;
    const int l16    = lane & 15;

    const int n0 = blockIdx.x * BN;
    const int h0 = blockIdx.y * BM;
    const int mtg = blockIdx.y * 4 + wave_m;     // global 16-row m-tile index
    const size_t bbase = (size_t)blockIdx.z * HDIM * NDIM;

    v8f acc[6][2];
#pragma unroll
    for (int s = 0; s < 6; ++s)
#pragma unroll
        for (int ns = 0; ns < 2; ++ns)
            acc[s][ns] = (v8f){0.f, 0.f, 0.f, 0.f, 0.f, 0.f, 0.f, 0.f};

    // ---- Prologue: stage B tile for k-step 0 into buffer 0 ----
    {
        f32x4 breg[4];
#pragma unroll
        for (int i = 0; i < 4; ++i) {
            int linear = i * 256 + tid;
            int mat = linear >> 9, rem = linear & 511, kk = rem >> 4, nc = rem & 15;
            const float* src = (mat == 0 ? m_wv : e_wv) + bbase + (size_t)kk * NDIM + n0 + nc * 4;
            breg[i] = *(const f32x4*)src;
        }
#pragma unroll
        for (int i = 0; i < 4; ++i) {
            int linear = i * 256 + tid;
            int mat = linear >> 9, rem = linear & 511, kk = rem >> 4, nc = rem & 15;
            int nb = nc * 4;
            Bs[(mat * BN + nb + 0) * PB + kk] = (f16)breg[i].x;
            Bs[(mat * BN + nb + 1) * PB + kk] = (f16)breg[i].y;
            Bs[(mat * BN + nb + 2) * PB + kk] = (f16)breg[i].z;
            Bs[(mat * BN + nb + 3) * PB + kk] = (f16)breg[i].w;
        }
        __syncthreads();
    }

    for (int k = 0; k < KSTEPS; ++k) {
        const int buf = k & 1;
        const int bufoff = buf * BBUF;
        const bool more = (k + 1) < KSTEPS;

        // ---- Issue global loads for k+1 BEFORE compute (latency hidden by WMMAs) ----
        f32x4 breg[4];
        if (more) {
            const int k0n = (k + 1) * BK;
#pragma unroll
            for (int i = 0; i < 4; ++i) {
                int linear = i * 256 + tid;
                int mat = linear >> 9, rem = linear & 511, kk = rem >> 4, nc = rem & 15;
                const float* src = (mat == 0 ? m_wv : e_wv) + bbase + (size_t)(k0n + kk) * NDIM + n0 + nc * 4;
                breg[i] = *(const f32x4*)src;
                if (k + 2 < KSTEPS)
                    __builtin_prefetch(src + (size_t)BK * NDIM, 0, 1);   // distance-2 prefetch
            }
        }

        // ---- B fragments from current LDS buffer ----
        V16U bf[2][2];
#pragma unroll
        for (int mat = 0; mat < 2; ++mat)
#pragma unroll
            for (int ns = 0; ns < 2; ++ns) {
                const f16* p = &Bs[bufoff + (mat * BN + wave_n * 32 + ns * 16 + l16) * PB + lhalf * 16];
                bf[mat][ns].h[0] = *(const f16x8*)(p);       // K = 16*kg .. +7
                bf[mat][ns].h[1] = *(const f16x8*)(p + 8);   // K = 16*kg+8 .. +15
            }

        // ---- A fragments straight from swizzled f16 weights (L2) + 12 WMMAs ----
#pragma unroll
        for (int s = 0; s < 6; ++s) {
            V16U af;
            const size_t cidx = ((((size_t)s * 32 + mtg) * 16 + k) * 32 + lane) * 16;
            af.h[0] = *(const f16x8*)(Wc + cidx);
            af.h[1] = *(const f16x8*)(Wc + cidx + 8);
            const int mat = (s < 3) ? 0 : 1;
#pragma unroll
            for (int ns = 0; ns < 2; ++ns)
                acc[s][ns] = __builtin_amdgcn_wmma_f32_16x16x32_f16(
                    false, af.v, false, bf[mat][ns].v,
                    (short)0, acc[s][ns], false, false);
        }

        // ---- Convert + store k+1 tile into the other buffer ----
        if (more) {
            const int aoff = (1 - buf) * BBUF;
#pragma unroll
            for (int i = 0; i < 4; ++i) {
                int linear = i * 256 + tid;
                int mat = linear >> 9, rem = linear & 511, kk = rem >> 4, nc = rem & 15;
                int nb = nc * 4;
                Bs[aoff + (mat * BN + nb + 0) * PB + kk] = (f16)breg[i].x;
                Bs[aoff + (mat * BN + nb + 1) * PB + kk] = (f16)breg[i].y;
                Bs[aoff + (mat * BN + nb + 2) * PB + kk] = (f16)breg[i].z;
                Bs[aoff + (mat * BN + nb + 3) * PB + kk] = (f16)breg[i].w;
            }
            __syncthreads();
        }
    }

    // ---- Fused GRU epilogue, all in registers ----
#pragma unroll
    for (int ns = 0; ns < 2; ++ns) {
        const int col = n0 + wave_n * 32 + ns * 16 + l16;
#pragma unroll
        for (int v = 0; v < 8; ++v) {
            const int row = h0 + wave_m * 16 + v + 8 * lhalf;   // C/D layout: VGPR v -> M = v + 8*(lane>>4)
            const size_t idx = bbase + (size_t)row * NDIM + col;
            const float e = e_wv[idx];
            float pre_r = acc[0][ns][v] + acc[3][ns][v];
            float pre_z = acc[1][ns][v] + acc[4][ns][v];
            float r = 1.f / (1.f + __expf(-pre_r));
            float z = 1.f / (1.f + __expf(-pre_z));
            float ng = tanhf(acc[2][ns][v] + r * acc[5][ns][v]);
            out[idx] = (1.f - z) * ng + z * e;
        }
    }
}

extern "C" void kernel_launch(void* const* d_in, const int* in_sizes, int n_in,
                              void* d_out, int out_size, void* d_ws, size_t ws_size,
                              hipStream_t stream) {
    const float* e_wv = (const float*)d_in[0];
    const float* m_wv = (const float*)d_in[1];
    const float* W_ih = (const float*)d_in[2];
    const float* W_hh = (const float*)d_in[3];
    float* out = (float*)d_out;
    f16* Wc = (f16*)d_ws;   // needs 2*1536*512*2 = 3 MiB of scratch

    convert_weights_kernel<<<768, 256, 0, stream>>>(W_ih, W_hh, Wc);

    dim3 grid(NDIM / BN, HDIM / BM, BATCH);   // (128, 8, 8)
    gru_fused_kernel<<<grid, 256, 0, stream>>>(e_wv, m_wv, Wc, out);
}
